// LSTMCell_78262894067860
// MI455X (gfx1250) — matrix operand order repaired and checked by reading, not compiled, not run on hardware
//
#include <hip/hip_runtime.h>
#include <hip/hip_bf16.h>
#include <math.h>
#include <stdint.h>

// ---------------------------------------------------------------------------
// LayerNorm-LSTM cell, MI455X (gfx1250, wave32, WMMA bf16 16x16x32, f32 acc)
//   B=8192, In=H=1024, K=In+H=2048, 4 gates
//   GEMM: per WG = 32 batch rows x full gate (1024 cols), K=2048
//   A tile -> LDS via Tensor Data Mover (double-buffered, TENSORcnt)
//   B stream -> registers, pair-pipelined 2 tiles ahead; all tile/chunk
//   offsets folded into load immediates (one base pointer per wave).
// ---------------------------------------------------------------------------

#define B_ROWS 8192
#define H_DIM  1024
#define K_DIM  2048
#define NGATE  4
#define LN_EPS 1e-6f

#define M_WG     32          // batch rows per workgroup (2 M-tiles per wave)
#define KPHASE   256         // K elements staged per TDM phase
#define NPHASE   (K_DIM / KPHASE)
#define KCHUNK   32          // K per WMMA
#define NCHUNK   (KPHASE / KCHUNK)
#define TILE_STRIDE (16 * K_DIM)   // elements between adjacent N-tiles of W

typedef __attribute__((ext_vector_type(16))) __bf16 bf16x16;
typedef __attribute__((ext_vector_type(4)))  __bf16 bf16x4;
typedef __attribute__((ext_vector_type(8)))  float  f32x8;
typedef unsigned int uint32x4 __attribute__((ext_vector_type(4)));
typedef int          int32x4  __attribute__((ext_vector_type(4)));
typedef int          int32x8  __attribute__((ext_vector_type(8)));

union FragBF {
    bf16x16 v;
    float4  q[2];   // two 16-byte chunks -> b128 loads
};

__device__ __forceinline__ float sigmoid_f(float x) {
    return 1.0f / (1.0f + __expf(-x));
}
__device__ __forceinline__ float tanh_f(float x) {
    return 1.0f - 2.0f / (__expf(2.0f * x) + 1.0f);
}

// ---------------------------------------------------------------------------
// TDM: tensor_load_to_lds of a (M_WG x KPHASE) bf16 tile from a row-major
// (.. x K_DIM) matrix into LDS (contiguous, KPHASE*2 bytes per row).
// Descriptor per CDNA5 ISA 8.3-8.5 (2D, data_size=2B, type=2).
// ---------------------------------------------------------------------------
__device__ __forceinline__ void tdm_load_tile(uint32_t lds_byte_addr,
                                              const __bf16* gptr /* tile start */)
{
    uint64_t ga = (uint64_t)(uintptr_t)gptr;

    uint32x4 g0;
    g0[0] = 1u;                                   // count=1, user descriptor
    g0[1] = lds_byte_addr;                        // lds_addr
    g0[2] = (uint32_t)(ga & 0xFFFFFFFFu);         // global_addr[31:0]
    g0[3] = (uint32_t)((ga >> 32) & 0x01FFFFFFu)  // global_addr[56:32]
          | (2u << 30);                           // type = 2 (image)

    int32x8 g1;
    g1[0] = (int)(1u << 16);                      // workgroup_mask=0, data_size=1 (2B)
    g1[1] = (int)(((uint32_t)K_DIM & 0xFFFFu) << 16);   // tensor_dim0[15:0]
    g1[2] = (int)((((uint32_t)K_DIM >> 16) & 0xFFFFu)
          | (((uint32_t)B_ROWS & 0xFFFFu) << 16));      // tensor_dim1[15:0]
    g1[3] = (int)(((uint32_t)B_ROWS >> 16) & 0xFFFFu)
          | (int)((uint32_t)KPHASE << 16);              // tile_dim0 = KPHASE
    g1[4] = (int)M_WG;                            // tile_dim1 = rows
    g1[5] = (int)K_DIM;                           // tensor_dim0_stride[31:0]
    g1[6] = 0;
    g1[7] = 0;

    int32x4 gz = {0, 0, 0, 0};

#if defined(__has_include)
#  if __has_include(<hip/amd_detail/amd_gfx1250_TDM.h>)
#    define ATHENA_TDM_6ARG 1
#  endif
#endif
#ifdef ATHENA_TDM_6ARG
    int32x8 gz8 = {0, 0, 0, 0, 0, 0, 0, 0};
    __builtin_amdgcn_tensor_load_to_lds(g0, g1, gz, gz, gz8, 0);
#else
    __builtin_amdgcn_tensor_load_to_lds(g0, g1, gz, gz, 0);
#endif
}

// ---------------------------------------------------------------------------
// Kernel 1a: pack A = [x | h] row-major (B_ROWS x K_DIM) bf16
// ---------------------------------------------------------------------------
__global__ void pack_A_kernel(const float* __restrict__ x,
                              const float* __restrict__ h,
                              __bf16* __restrict__ A)
{
    int idx = blockIdx.x * blockDim.x + threadIdx.x;
    int total = B_ROWS * (H_DIM / 4);
    if (idx >= total) return;
    int b  = idx / (H_DIM / 4);
    int j4 = (idx % (H_DIM / 4)) * 4;

    float4 xv = *reinterpret_cast<const float4*>(x + (size_t)b * H_DIM + j4);
    float4 hv = *reinterpret_cast<const float4*>(h + (size_t)b * H_DIM + j4);

    bf16x4 xa, ha;
    xa[0] = (__bf16)xv.x; xa[1] = (__bf16)xv.y; xa[2] = (__bf16)xv.z; xa[3] = (__bf16)xv.w;
    ha[0] = (__bf16)hv.x; ha[1] = (__bf16)hv.y; ha[2] = (__bf16)hv.z; ha[3] = (__bf16)hv.w;

    __bf16* row = A + (size_t)b * K_DIM;
    *reinterpret_cast<bf16x4*>(row + j4)         = xa;
    *reinterpret_cast<bf16x4*>(row + H_DIM + j4) = ha;
}

// ---------------------------------------------------------------------------
// Kernel 1b: pack W = [w_ih | w_hh] row-major (4H x K_DIM) bf16
// ---------------------------------------------------------------------------
__global__ void pack_W_kernel(const float* __restrict__ w_ih,
                              const float* __restrict__ w_hh,
                              __bf16* __restrict__ W)
{
    int idx = blockIdx.x * blockDim.x + threadIdx.x;
    int total = NGATE * H_DIM * (H_DIM / 4);
    if (idx >= total) return;
    int k  = idx / (H_DIM / 4);
    int j4 = (idx % (H_DIM / 4)) * 4;

    float4 av = *reinterpret_cast<const float4*>(w_ih + (size_t)k * H_DIM + j4);
    float4 bv = *reinterpret_cast<const float4*>(w_hh + (size_t)k * H_DIM + j4);

    bf16x4 aa, ba;
    aa[0] = (__bf16)av.x; aa[1] = (__bf16)av.y; aa[2] = (__bf16)av.z; aa[3] = (__bf16)av.w;
    ba[0] = (__bf16)bv.x; ba[1] = (__bf16)bv.y; ba[2] = (__bf16)bv.z; ba[3] = (__bf16)bv.w;

    __bf16* row = W + (size_t)k * K_DIM;
    *reinterpret_cast<bf16x4*>(row + j4)         = aa;
    *reinterpret_cast<bf16x4*>(row + H_DIM + j4) = ba;
}

// ---------------------------------------------------------------------------
// Kernel 2: fused GEMM (v = A*W^T + b) + per-gate LayerNorm + activation.
// Grid: (B_ROWS/M_WG, 4). Block: 256 (= 8 wave32).
// ---------------------------------------------------------------------------
__global__ __launch_bounds__(256) void gemm_ln_gate_kernel(
        const __bf16* __restrict__ A,       // (B_ROWS x K_DIM)
        const __bf16* __restrict__ W,       // (4H x K_DIM)
        const float*  __restrict__ b_ih,
        const float*  __restrict__ gamma,
        const float*  __restrict__ beta,
        float*        __restrict__ gates)   // (B_ROWS x 4H) activated
{
    const int m0   = blockIdx.x * M_WG;
    const int g    = blockIdx.y;
    const int tid  = threadIdx.x;
    const int wave = tid >> 5;
    const int lane = tid & 31;
    const int half = lane >> 4;
    const int l16  = lane & 15;
    const int colBase = wave * 128;

    __shared__ __bf16 sA[2][M_WG * KPHASE];   // 2 x 16 KB
    __shared__ float  s_sum[M_WG];
    __shared__ float  s_sq[M_WG];

    if (tid < M_WG) { s_sum[tid] = 0.f; s_sq[tid] = 0.f; }

    // ---- accumulators: 2 M-tiles x 8 N-tiles ----
    f32x8 acc[16];
    const f32x8 zero8 = {0.f, 0.f, 0.f, 0.f, 0.f, 0.f, 0.f, 0.f};
    #pragma unroll
    for (int t = 0; t < 16; ++t) acc[t] = zero8;

    // Single per-lane B base pointer; tile/chunk offsets are constants that
    // fold into the 24-bit global-load immediate. Advanced once per phase.
    const int kbB = half * 16;                // B frag: K {16*half..16*half+15}
    const __bf16* bbase =
        W + (size_t)(g * H_DIM + colBase + l16) * K_DIM + kbB;

    const int kbA = half * 16;                // A frag: K {16*half..16*half+15}

    // ---- prologue: TDM-load phase 0 ----
    if (wave == 0) {
        tdm_load_tile((uint32_t)(uintptr_t)&sA[0][0],
                      A + (size_t)m0 * K_DIM);
        __builtin_amdgcn_s_wait_tensorcnt(0);
    }
    __syncthreads();

    for (int ph = 0; ph < NPHASE; ++ph) {
        const int cur = ph & 1;
        if (wave == 0 && ph + 1 < NPHASE) {
            tdm_load_tile((uint32_t)(uintptr_t)&sA[cur ^ 1][0],
                          A + (size_t)m0 * K_DIM + (size_t)(ph + 1) * KPHASE);
        }

        // per-phase A base in LDS; chunk/row offsets fold into ds immediates
        const __bf16* sAbase = &sA[cur][l16 * KPHASE + kbA];

        #pragma unroll
        for (int c = 0; c < NCHUNK; ++c) {
            const int kl = c * KCHUNK;        // unroll-constant chunk offset

            // prefetch B one phase ahead (rotating tile) -> global_prefetch_b8
            if (ph + 1 < NPHASE)
                __builtin_prefetch(bbase + (size_t)(c & 7) * TILE_STRIDE + kl + KPHASE, 0, 3);

            // A fragments for both M-tiles from LDS (ds_load_b128 x2 each)
            FragBF a0, a1;
            a0.q[0] = *reinterpret_cast<const float4*>(sAbase + kl);
            a0.q[1] = *reinterpret_cast<const float4*>(sAbase + kl + 8);
            a1.q[0] = *reinterpret_cast<const float4*>(sAbase + 16 * KPHASE + kl);
            a1.q[1] = *reinterpret_cast<const float4*>(sAbase + 16 * KPHASE + kl + 8);

            // B fragments: pair-pipelined, 2 tiles of lookahead
            FragBF f0, f1, n0, n1;
            f0.q[0] = *reinterpret_cast<const float4*>(bbase + 0 * TILE_STRIDE + kl);
            f0.q[1] = *reinterpret_cast<const float4*>(bbase + 0 * TILE_STRIDE + kl + 8);
            f1.q[0] = *reinterpret_cast<const float4*>(bbase + 1 * TILE_STRIDE + kl);
            f1.q[1] = *reinterpret_cast<const float4*>(bbase + 1 * TILE_STRIDE + kl + 8);

            #pragma unroll
            for (int p = 0; p < 4; ++p) {
                if (p < 3) {
                    const size_t o0 = (size_t)(2 * p + 2) * TILE_STRIDE + kl;
                    const size_t o1 = (size_t)(2 * p + 3) * TILE_STRIDE + kl;
                    n0.q[0] = *reinterpret_cast<const float4*>(bbase + o0);
                    n0.q[1] = *reinterpret_cast<const float4*>(bbase + o0 + 8);
                    n1.q[0] = *reinterpret_cast<const float4*>(bbase + o1);
                    n1.q[1] = *reinterpret_cast<const float4*>(bbase + o1 + 8);
                }
                const int t0 = 2 * p, t1 = 2 * p + 1;
                acc[t0]     = __builtin_amdgcn_wmma_f32_16x16x32_bf16(
                                  false, a0.v, false, f0.v, (short)0, acc[t0],     false, false);
                acc[8 + t0] = __builtin_amdgcn_wmma_f32_16x16x32_bf16(
                                  false, a1.v, false, f0.v, (short)0, acc[8 + t0], false, false);
                acc[t1]     = __builtin_amdgcn_wmma_f32_16x16x32_bf16(
                                  false, a0.v, false, f1.v, (short)0, acc[t1],     false, false);
                acc[8 + t1] = __builtin_amdgcn_wmma_f32_16x16x32_bf16(
                                  false, a1.v, false, f1.v, (short)0, acc[8 + t1], false, false);
                f0 = n0; f1 = n1;
            }
        }

        bbase += KPHASE;   // single pointer advance per phase

        if (wave == 0 && ph + 1 < NPHASE)
            __builtin_amdgcn_s_wait_tensorcnt(0);
        __syncthreads();
    }

    // ---- bias + LayerNorm over each gate row ----
    float bias[8], gam[8], bet[8];
    #pragma unroll
    for (int t = 0; t < 8; ++t) {
        int col = colBase + t * 16 + l16;
        bias[t] = b_ih [g * H_DIM + col];
        gam[t]  = gamma[g * H_DIM + col];
        bet[t]  = beta [g * H_DIM + col];
    }
    #pragma unroll
    for (int t = 0; t < 16; ++t)
        #pragma unroll
        for (int j = 0; j < 8; ++j) acc[t][j] += bias[t & 7];

    // C/D layout: VGPR j holds row (mt*16 + j + 8*half), col = l16 + tile*16
    #pragma unroll
    for (int mt = 0; mt < 2; ++mt) {
        #pragma unroll
        for (int j = 0; j < 8; ++j) {
            float ps = 0.f, pq = 0.f;
            #pragma unroll
            for (int t = 0; t < 8; ++t) {
                float v = acc[mt * 8 + t][j];
                ps += v; pq += v * v;
            }
            for (int m = 1; m < 16; m <<= 1) {
                ps += __shfl_xor(ps, m, 32);
                pq += __shfl_xor(pq, m, 32);
            }
            if (l16 == 0) {
                int row = mt * 16 + j + half * 8;
                atomicAdd(&s_sum[row], ps);    // ds_add_f32
                atomicAdd(&s_sq[row],  pq);
            }
        }
    }
    __syncthreads();

    #pragma unroll
    for (int mt = 0; mt < 2; ++mt) {
        #pragma unroll
        for (int j = 0; j < 8; ++j) {
            int   row = mt * 16 + j + half * 8;
            float mu  = s_sum[row] * (1.0f / H_DIM);
            float var = (s_sq[row] - (float)H_DIM * mu * mu) * (1.0f / (H_DIM - 1));
            float inv = 1.0f / (sqrtf(fmaxf(var, 0.f)) + LN_EPS);
            #pragma unroll
            for (int t = 0; t < 8; ++t) {
                float y = gam[t] * (acc[mt * 8 + t][j] - mu) * inv + bet[t];
                float a;
                if      (g == 1) a = sigmoid_f(y + 1.0f);
                else if (g == 2) a = tanh_f(y);
                else             a = sigmoid_f(y);
                int col = colBase + t * 16 + l16;
                gates[(size_t)(m0 + row) * (NGATE * H_DIM) + g * H_DIM + col] = a;
            }
        }
    }
}

// ---------------------------------------------------------------------------
// Kernel 3: new_c = LN(c*f + i*g), new_h = tanh(new_c)*o.
// ---------------------------------------------------------------------------
__global__ __launch_bounds__(256) void cell_update_kernel(
        const float* __restrict__ gates,
        const float* __restrict__ c,
        const float* __restrict__ gamma_c,
        const float* __restrict__ beta_c,
        float*       __restrict__ out)
{
    const int b   = blockIdx.x;
    const int tid = threadIdx.x;
    const float* gp = gates + (size_t)b * (NGATE * H_DIM);

    float pre[4], og[4];
    float ps = 0.f, pq = 0.f;
    #pragma unroll
    for (int u = 0; u < 4; ++u) {
        int col  = tid + u * 256;
        float iv = gp[0 * H_DIM + col];
        float fv = gp[1 * H_DIM + col];
        float gv = gp[2 * H_DIM + col];
        float ov = gp[3 * H_DIM + col];
        float cv = c[(size_t)b * H_DIM + col];
        float p  = cv * fv + iv * gv;
        pre[u] = p; og[u] = ov;
        ps += p; pq += p * p;
    }
    for (int m = 1; m < 32; m <<= 1) {
        ps += __shfl_xor(ps, m, 32);
        pq += __shfl_xor(pq, m, 32);
    }
    __shared__ float sw[8], sq[8];
    __shared__ float s_mu, s_inv;
    int wave = tid >> 5, lane = tid & 31;
    if (lane == 0) { sw[wave] = ps; sq[wave] = pq; }
    __syncthreads();
    if (tid == 0) {
        float S = 0.f, Q = 0.f;
        for (int w = 0; w < 8; ++w) { S += sw[w]; Q += sq[w]; }
        float mu  = S * (1.0f / H_DIM);
        float var = (Q - (float)H_DIM * mu * mu) * (1.0f / (H_DIM - 1));
        s_mu  = mu;
        s_inv = 1.0f / (sqrtf(fmaxf(var, 0.f)) + LN_EPS);
    }
    __syncthreads();

    #pragma unroll
    for (int u = 0; u < 4; ++u) {
        int col  = tid + u * 256;
        float nc = gamma_c[col] * (pre[u] - s_mu) * s_inv + beta_c[col];
        float nh = tanh_f(nc) * og[u];
        out[(size_t)b * H_DIM + col] = nh;
        out[(size_t)B_ROWS * H_DIM + (size_t)b * H_DIM + col] = nc;
    }
}

// ---------------------------------------------------------------------------
// Host-side launcher
// ---------------------------------------------------------------------------
extern "C" void kernel_launch(void* const* d_in, const int* in_sizes, int n_in,
                              void* d_out, int out_size, void* d_ws, size_t ws_size,
                              hipStream_t stream) {
    const float* x          = (const float*)d_in[0];
    const float* h          = (const float*)d_in[1];
    const float* c          = (const float*)d_in[2];
    const float* w_ih       = (const float*)d_in[3];
    const float* b_ih       = (const float*)d_in[4];
    const float* w_hh       = (const float*)d_in[5];
    const float* gamma_ifgo = (const float*)d_in[6];
    const float* beta_ifgo  = (const float*)d_in[7];
    const float* gamma_c    = (const float*)d_in[8];
    const float* beta_c     = (const float*)d_in[9];
    float* out = (float*)d_out;

    char* ws = (char*)d_ws;
    const size_t A_bytes = (size_t)B_ROWS * K_DIM * sizeof(__bf16);
    const size_t W_bytes = (size_t)NGATE * H_DIM * K_DIM * sizeof(__bf16);
    __bf16* Abf     = (__bf16*)ws;
    __bf16* Wbf     = (__bf16*)(ws + A_bytes);
    float*  gatebuf = (float*)(ws + A_bytes + W_bytes);

    {
        int totalA = B_ROWS * (H_DIM / 4);
        pack_A_kernel<<<(totalA + 255) / 256, 256, 0, stream>>>(x, h, Abf);
        int totalW = NGATE * H_DIM * (H_DIM / 4);
        pack_W_kernel<<<(totalW + 255) / 256, 256, 0, stream>>>(w_ih, w_hh, Wbf);
    }
    {
        dim3 grid(B_ROWS / M_WG, NGATE);
        gemm_ln_gate_kernel<<<grid, 256, 0, stream>>>(
            Abf, Wbf, b_ih, gamma_ifgo, beta_ifgo, gatebuf);
    }
    cell_update_kernel<<<B_ROWS, 256, 0, stream>>>(
        gatebuf, c, gamma_c, beta_c, out);
}
